// MoBAAttention_19378892439586
// MI455X (gfx1250) — matrix operand order repaired and checked by reading, not compiled, not run on hardware
//
#include <hip/hip_runtime.h>
#include <hip/hip_bf16.h>

// ---------------------------------------------------------------------------
// MoBA attention for MI455X (gfx1250, wave32, WMMA 16x16x32 f16 -> f32 acc)
// H=1024, NH=16, HD=64, B=2, S=4096, scales bs = 1024,512,256,128
// GEMMs use 32x32 register tiles per wave: 4 WMMAs per K-step, 2 b128/WMMA.
// ---------------------------------------------------------------------------

typedef __attribute__((ext_vector_type(16))) _Float16 v16h;
typedef __attribute__((ext_vector_type(8)))  _Float16 v8h;
typedef __attribute__((ext_vector_type(8)))  float    v8f;

#define H_DIM   1024
#define NH      16
#define HD      64
#define BATCH   2
#define SEQ     4096
#define ROWS    (BATCH * SEQ)          // 8192
#define SCALEF  0.125f                 // 1/sqrt(64)
#define NEGF    (-3.0e38f)

// ---- WMMA wrapper ----------------------------------------------------------
__device__ __forceinline__ v8f wmma16(v16h a, v16h b, v8f c) {
  return __builtin_amdgcn_wmma_f32_16x16x32_f16(
      /*neg_a=*/false, a, /*neg_b=*/false, b,
      /*c_mod=*/(short)0, c, /*reuse_a=*/false, /*reuse_b=*/false);
}

// ---- A-matrix 16x32 (f16) load, row-major src, row stride ld ---------------
// lanes 0-15: row=lane, h[0..7]=K 0..7, h[8..15]=K 16..23
// lanes 16-31: row=lane-16, h[0..7]=K 8..15, h[8..15]=K 24..31
__device__ __forceinline__ v16h load_a16x32(const _Float16* p, int ld, int lane) {
  const int r  = lane & 15;
  const int hf = (lane >> 4) & 1;
  const _Float16* row = p + r * ld + hf * 8;
  v8h lo = *(const v8h*)(row);
  v8h hi = *(const v8h*)(row + 16);
  return __builtin_shufflevector(lo, hi, 0,1,2,3,4,5,6,7,8,9,10,11,12,13,14,15);
}

// ---- B-matrix 32x16 load where memory holds W as (N,K) row-major ----------
// B[k][n] = W[n][k].  lanes 0-15: n=lane, K 0..7,16..23; lanes 16-31: K 8..15,24..31.
__device__ __forceinline__ v16h load_bt16(const _Float16* w, int ld, int lane) {
  const int r  = lane & 15;
  const int hf = (lane >> 4) & 1;
  const _Float16* row = w + r * ld + hf * 8;
  v8h lo = *(const v8h*)(row);
  v8h hi = *(const v8h*)(row + 16);
  return __builtin_shufflevector(lo, hi, 0,1,2,3,4,5,6,7,8,9,10,11,12,13,14,15);
}

// ---- row reductions within 16-lane half (C-matrix rows live per half) ------
__device__ __forceinline__ float rmax16(float v) {
  v = fmaxf(v, __shfl_xor(v, 1, 32));
  v = fmaxf(v, __shfl_xor(v, 2, 32));
  v = fmaxf(v, __shfl_xor(v, 4, 32));
  v = fmaxf(v, __shfl_xor(v, 8, 32));
  return v;
}
__device__ __forceinline__ float rsum16(float v) {
  v += __shfl_xor(v, 1, 32);
  v += __shfl_xor(v, 2, 32);
  v += __shfl_xor(v, 4, 32);
  v += __shfl_xor(v, 8, 32);
  return v;
}

// ---------------------------------------------------------------------------
// f32 -> f16 conversion
// ---------------------------------------------------------------------------
__global__ __launch_bounds__(256)
void moba_cvt_kernel(const float* __restrict__ src, _Float16* __restrict__ dst, int n) {
  int i = blockIdx.x * 256 + threadIdx.x;
  if (i < n) dst[i] = (_Float16)src[i];
}

// ---------------------------------------------------------------------------
// Fused QKV projection: q/k/v = x @ w{q,k,v}.T ; q,k *= -sin ; v transposed.
// One wave per 32x32 output tile (2x2 WMMA subtiles, operands reused).
// ---------------------------------------------------------------------------
__global__ __launch_bounds__(256)
void moba_qkv_kernel(const _Float16* __restrict__ xh,
                     const _Float16* __restrict__ wqh,
                     const _Float16* __restrict__ wkh,
                     const _Float16* __restrict__ wvh,
                     const float*    __restrict__ sinp,
                     _Float16* __restrict__ qh,   // (bh, s, d)
                     _Float16* __restrict__ kh,   // (bh, s, d)
                     _Float16* __restrict__ vT) { // (bh, d, s)
  const int lane = threadIdx.x & 31;
  const int wid  = blockIdx.x * 8 + (threadIdx.x >> 5);
  const int TPM  = (ROWS / 32) * (H_DIM / 32);      // 8192 tiles / matrix
  const int mat  = wid / TPM;
  const int t    = wid % TPM;
  const int m0   = (t >> 5) << 5;                   // 256 row tiles of 32
  const int n0   = (t & 31) << 5;                   // 32 col tiles of 32
  const _Float16* w = (mat == 0) ? wqh : (mat == 1) ? wkh : wvh;

  v8f c00 = {}, c01 = {}, c10 = {}, c11 = {};
  for (int k0 = 0; k0 < H_DIM; k0 += 32) {
    __builtin_prefetch(xh + m0 * H_DIM + k0 + 128, 0, 1);
    const v16h a0 = load_a16x32(xh + m0 * H_DIM + k0,        H_DIM, lane);
    const v16h a1 = load_a16x32(xh + (m0 + 16) * H_DIM + k0, H_DIM, lane);
    const v16h b0 = load_bt16(w + n0 * H_DIM + k0,           H_DIM, lane);
    const v16h b1 = load_bt16(w + (n0 + 16) * H_DIM + k0,    H_DIM, lane);
    c00 = wmma16(a0, b0, c00);
    c01 = wmma16(a0, b1, c01);
    c10 = wmma16(a1, b0, c10);
    c11 = wmma16(a1, b1, c11);
  }

  const int r  = lane & 15;
  const int hf = lane >> 4;
  v8f ct[2][2] = {{c00, c01}, {c10, c11}};
  for (int si = 0; si < 2; ++si) {
    for (int sj = 0; sj < 2; ++sj) {
      const int col = n0 + sj * 16 + r;
      const int hh  = col >> 6;
      const int d   = col & 63;
      for (int i = 0; i < 8; ++i) {
        const int m  = m0 + si * 16 + hf * 8 + i;   // C layout: M = i + 8*half
        const int b  = m >> 12;                     // / 4096
        const int s  = m & (SEQ - 1);
        const int bh = b * NH + hh;
        const float val = ct[si][sj][i];
        if (mat == 2) {
          vT[(size_t)(bh * HD + d) * SEQ + s] = (_Float16)val;
        } else {
          const float sv = -sinp[s * HD + d];
          _Float16* dst = (mat == 0) ? qh : kh;
          dst[((size_t)bh * SEQ + s) * HD + d] = (_Float16)(val * sv);
        }
      }
    }
  }
}

// ---------------------------------------------------------------------------
// Multi-scale block attention (flash-style).  One wave per (bh, 16-query tile).
// Every scale attends to the FIRST bs keys with mask  key <= (q mod bs)+row.
// ---------------------------------------------------------------------------
__global__ __launch_bounds__(256)
void moba_attn_kernel(const _Float16* __restrict__ qh,
                      const _Float16* __restrict__ kh,
                      const _Float16* __restrict__ vT,
                      _Float16* __restrict__ ctxh) {  // (B,S,H) f16
  __shared__ __align__(16) _Float16 pbuf[8][16 * 32];

  const int lane = threadIdx.x & 31;
  const int warp = threadIdx.x >> 5;
  const int wid  = blockIdx.x * 8 + warp;         // 0 .. 8191
  const int bh   = wid >> 8;                      // 32 (b,h) pairs
  const int q0   = (wid & 255) << 4;              // query tile start
  const int r    = lane & 15;
  const int hf   = lane >> 4;

  const _Float16* qb = qh + ((size_t)bh * SEQ + q0) * HD;
  const _Float16* kb = kh + (size_t)bh * SEQ * HD;
  const _Float16* vb = vT + (size_t)bh * HD * SEQ;
  _Float16* pw = pbuf[warp];

  // Q A-operand tiles (16 rows x 64 dims -> two 16x32 chunks), loaded once.
  const v16h qa0 = load_a16x32(qb, HD, lane);
  const v16h qa1 = load_a16x32(qb + 32, HD, lane);

  const v8f vzero = {};
  v8f tot[4] = {vzero, vzero, vzero, vzero};

  for (int sc = 0; sc < 4; ++sc) {
    const int bs    = 1024 >> sc;
    const int qmod  = q0 & (bs - 1);
    const int nkeys = qmod + 16;                  // keys 0 .. qmod+15 used

    float mi[8], li[8];
    for (int i = 0; i < 8; ++i) { mi[i] = NEGF; li[i] = 0.0f; }
    v8f acc[4] = {vzero, vzero, vzero, vzero};

    for (int k0 = 0; k0 < nkeys; k0 += 32) {
      // ---- scores: two 16x16 tiles (keys k0..k0+15, k0+16..k0+31) ----
      v8f s0 = wmma16(qa0, load_bt16(kb + k0 * HD,             HD, lane), vzero);
      s0     = wmma16(qa1, load_bt16(kb + k0 * HD + 32,        HD, lane), s0);
      v8f s1 = wmma16(qa0, load_bt16(kb + (k0 + 16) * HD,      HD, lane), vzero);
      s1     = wmma16(qa1, load_bt16(kb + (k0 + 16) * HD + 32, HD, lane), s1);

      // ---- online softmax (half 0 -> rows 0..7, half 1 -> rows 8..15) ----
      for (int i = 0; i < 8; ++i) {
        const int rq = qmod + hf * 8 + i;         // q position mod bs
        const float f0 = (k0 + r      <= rq) ? s0[i] * SCALEF : NEGF;
        const float f1 = (k0 + 16 + r <= rq) ? s1[i] * SCALEF : NEGF;
        const float mx = rmax16(fmaxf(f0, f1));
        const float mn = fmaxf(mi[i], mx);
        const float alpha = __expf(mi[i] - mn);
        const float p0 = __expf(f0 - mn);
        const float p1 = __expf(f1 - mn);
        li[i] = li[i] * alpha + rsum16(p0 + p1);
        mi[i] = mn;
        for (int tt = 0; tt < 4; ++tt) acc[tt][i] *= alpha;
        const int row = hf * 8 + i;               // C layout -> LDS row-major
        pw[row * 32 + r]      = (_Float16)p0;
        pw[row * 32 + 16 + r] = (_Float16)p1;
      }

      // ---- ctx += P(16x32) @ V(32x64): 4 WMMAs over HD column tiles ----
      // same-wave LDS store->load is in-order; no barrier needed
      const v16h pa = load_a16x32(pw, 32, lane);
      for (int tt = 0; tt < 4; ++tt)
        acc[tt] = wmma16(pa, load_bt16(vb + (tt * 16) * SEQ + k0, SEQ, lane), acc[tt]);
    }

    for (int i = 0; i < 8; ++i) {
      const float inv = 0.25f / li[i];
      for (int tt = 0; tt < 4; ++tt) tot[tt][i] += acc[tt][i] * inv;
    }
  }

  // store ctx in (B,S,H) layout as f16 for the output projection GEMM
  const int b  = bh >> 4;
  const int hh = bh & 15;
  for (int i = 0; i < 8; ++i) {
    const int s = q0 + hf * 8 + i;
    for (int tt = 0; tt < 4; ++tt) {
      ctxh[((size_t)(b * SEQ + s)) * H_DIM + hh * HD + tt * 16 + r] =
          (_Float16)tot[tt][i];
    }
  }
}

// ---------------------------------------------------------------------------
// Output projection: out = ctx @ wo.T (f32 result), 32x32 tile per wave
// ---------------------------------------------------------------------------
__global__ __launch_bounds__(256)
void moba_oproj_kernel(const _Float16* __restrict__ ctxh,
                       const _Float16* __restrict__ woh,
                       float* __restrict__ out) {
  const int lane = threadIdx.x & 31;
  const int wid  = blockIdx.x * 8 + (threadIdx.x >> 5);
  const int m0   = (wid >> 5) << 5;
  const int n0   = (wid & 31) << 5;

  v8f c00 = {}, c01 = {}, c10 = {}, c11 = {};
  for (int k0 = 0; k0 < H_DIM; k0 += 32) {
    __builtin_prefetch(ctxh + m0 * H_DIM + k0 + 128, 0, 1);
    const v16h a0 = load_a16x32(ctxh + m0 * H_DIM + k0,        H_DIM, lane);
    const v16h a1 = load_a16x32(ctxh + (m0 + 16) * H_DIM + k0, H_DIM, lane);
    const v16h b0 = load_bt16(woh + n0 * H_DIM + k0,           H_DIM, lane);
    const v16h b1 = load_bt16(woh + (n0 + 16) * H_DIM + k0,    H_DIM, lane);
    c00 = wmma16(a0, b0, c00);
    c01 = wmma16(a0, b1, c01);
    c10 = wmma16(a1, b0, c10);
    c11 = wmma16(a1, b1, c11);
  }

  const int r  = lane & 15;
  const int hf = lane >> 4;
  v8f ct[2][2] = {{c00, c01}, {c10, c11}};
  for (int si = 0; si < 2; ++si)
    for (int sj = 0; sj < 2; ++sj)
      for (int i = 0; i < 8; ++i)
        out[(size_t)(m0 + si * 16 + hf * 8 + i) * H_DIM + n0 + sj * 16 + r] =
            ct[si][sj][i];
}

// ---------------------------------------------------------------------------
// Host-side launch sequence
// ---------------------------------------------------------------------------
extern "C" void kernel_launch(void* const* d_in, const int* in_sizes, int n_in,
                              void* d_out, int out_size, void* d_ws, size_t ws_size,
                              hipStream_t stream) {
  (void)n_in; (void)out_size; (void)ws_size; (void)in_sizes;
  const float* x    = (const float*)d_in[0];
  const float* wq   = (const float*)d_in[1];
  const float* wk   = (const float*)d_in[2];
  const float* wv   = (const float*)d_in[3];
  const float* wo   = (const float*)d_in[4];
  // d_in[5] = cos (unused by the reference math)
  const float* sinp = (const float*)d_in[6];

  char* ws = (char*)d_ws;
  size_t off = 0;
  const size_t XSZ = (size_t)ROWS * H_DIM;        // 8388608 elems
  const size_t WSZ = (size_t)H_DIM * H_DIM;       // 1048576 elems

  _Float16* xh  = (_Float16*)(ws + off); off += XSZ * 2;
  _Float16* wqh = (_Float16*)(ws + off); off += WSZ * 2;
  _Float16* wkh = (_Float16*)(ws + off); off += WSZ * 2;
  _Float16* wvh = (_Float16*)(ws + off); off += WSZ * 2;
  _Float16* woh = (_Float16*)(ws + off); off += WSZ * 2;
  _Float16* qh  = (_Float16*)(ws + off); off += XSZ * 2;
  _Float16* kh  = (_Float16*)(ws + off); off += XSZ * 2;
  _Float16* vT  = (_Float16*)(ws + off); off += XSZ * 2;
  _Float16* ctx = (_Float16*)(ws + off); off += XSZ * 2;

  moba_cvt_kernel<<<(int)((XSZ + 255) / 256), 256, 0, stream>>>(x,  xh,  (int)XSZ);
  moba_cvt_kernel<<<(int)((WSZ + 255) / 256), 256, 0, stream>>>(wq, wqh, (int)WSZ);
  moba_cvt_kernel<<<(int)((WSZ + 255) / 256), 256, 0, stream>>>(wk, wkh, (int)WSZ);
  moba_cvt_kernel<<<(int)((WSZ + 255) / 256), 256, 0, stream>>>(wv, wvh, (int)WSZ);
  moba_cvt_kernel<<<(int)((WSZ + 255) / 256), 256, 0, stream>>>(wo, woh, (int)WSZ);

  // 3 matrices * 8192 32x32 tiles / 8 waves per block = 3072 blocks
  moba_qkv_kernel<<<3072, 256, 0, stream>>>(xh, wqh, wkh, wvh, sinp, qh, kh, vT);

  // 32 (b,h) * 256 q-tiles / 8 waves per block = 1024 blocks
  moba_attn_kernel<<<1024, 256, 0, stream>>>(qh, kh, vT, ctx);

  // 8192 32x32 tiles / 8 waves per block = 1024 blocks
  moba_oproj_kernel<<<1024, 256, 0, stream>>>(ctx, woh, (float*)d_out);
}
